// InputEmbedding_36661840839340
// MI455X (gfx1250) — matrix operand order repaired and checked by reading, not compiled, not run on hardware
//
#include <hip/hip_runtime.h>
#include <hip/hip_bf16.h>

typedef __attribute__((ext_vector_type(16))) _Float16 v16h;
typedef __attribute__((ext_vector_type(8)))  _Float16 v8h;
typedef __attribute__((ext_vector_type(8)))  float    v8f;

#define NPT   4096
#define BATCH 4
#define NS    32
#define MTOT  (BATCH * NPT * NS)   // 524288 positions
#define RAD2  0.01f
#define EPSBN 1e-5f

// ---------------------------------------------------------------------------
// Feature emit: f16, 32 channels per position (9 real + zero pad) so the
// GEMM B-fragment load is a contiguous 32-byte vector per lane.
// ---------------------------------------------------------------------------
__device__ __forceinline__ void emit_feat(_Float16* __restrict__ X1, size_t pos,
                                          float gx, float gy, float gz,
                                          float qx, float qy, float qz) {
  v16h v = {};
  v[0] = (_Float16)(gx - qx); v[1] = (_Float16)(gy - qy); v[2] = (_Float16)(gz - qz);
  v[3] = (_Float16)qx;        v[4] = (_Float16)qy;        v[5] = (_Float16)qz;
  v[6] = (_Float16)gx;        v[7] = (_Float16)gy;        v[8] = (_Float16)gz;
  v16h z = {};
  v16h* p = (v16h*)(X1 + pos * 32);
  p[0] = v;
  p[1] = z;
}

// ---------------------------------------------------------------------------
// Ball query + feature build, one WAVE per query point (16384 waves total).
// Batch's 48 KB of points staged in LDS; 32 lanes test 32 candidates/step in
// index order. Ballot mask order == index order, so popc of the lower-lane
// bits gives the output slot -- exact match of the reference's
// sort-then-take-first-32 semantics. Early exit once 32 hits are found.
// ---------------------------------------------------------------------------
__global__ __launch_bounds__(256) void ball_query_feats(
    const float* __restrict__ xyz, _Float16* __restrict__ X1,
    float* __restrict__ out_xyz) {
  __shared__ float sx[NPT], sy[NPT], sz[NPT];
  const int b = blockIdx.y;
  const float* base = xyz + (size_t)b * NPT * 3;
  for (int i = threadIdx.x; i < NPT; i += 256) {
    sx[i] = base[3 * i + 0];
    sy[i] = base[3 * i + 1];
    sz[i] = base[3 * i + 2];
  }
  __syncthreads();

  const int wv   = threadIdx.x >> 5;
  const int lane = threadIdx.x & 31;
  const int n    = blockIdx.x * 8 + wv;        // this wave's query point
  const float qx = sx[n], qy = sy[n], qz = sz[n];
  const size_t onb = (size_t)(b * NPT + n);
  if (lane == 0) {                              // xyz passthrough
    out_xyz[onb * 3 + 0] = qx;
    out_xyz[onb * 3 + 1] = qy;
    out_xyz[onb * 3 + 2] = qz;
  }
  const size_t pbase = onb * NS;

  int  cnt    = 0;
  int  firstm = n;          // self is always within radius; safe default
  bool haveFirst = false;
  for (int basem = 0; basem < NPT && cnt < NS; basem += 32) {
    const int m = basem + lane;
    const float dx = sx[m] - qx, dy = sy[m] - qy, dz = sz[m] - qz;
    const bool hit = (dx * dx + dy * dy + dz * dz) <= RAD2;
    const unsigned long long mask = __ballot(hit);
    if (!haveFirst && mask != 0ull) {
      firstm = basem + (int)__builtin_ctzll(mask);
      haveFirst = true;
    }
    const int pos = cnt + (int)__popcll(mask & ((1ull << lane) - 1ull));
    if (hit && pos < NS)
      emit_feat(X1, pbase + pos, sx[m], sy[m], sz[m], qx, qy, qz);
    cnt += (int)__popcll(mask);
  }
  if (cnt < NS) {           // pad remaining slots with the first hit, 1/lane
    const float fx = sx[firstm], fy = sy[firstm], fz = sz[firstm];
    const int s = cnt + lane;
    if (s < NS) emit_feat(X1, pbase + s, fx, fy, fz, qx, qy, qz);
  }
}

// ---------------------------------------------------------------------------
// WMMA GEMM (K, O compile-time): Y[O, MTOT] (f16) = Wh[O,K] x X[MTOT,K]^T
// + bias, fused per-channel sum/sumsq for BatchNorm.
// Each wave: one 16-channel tile x 4 consecutive 16-column tiles.
// All B fragments loaded up front (vmem clause, overlapped with the WMMA
// chain); stats accumulated in registers across all 4 tiles and reduced once
// per wave (shuffle -> LDS atomics -> 1 global atomic/ch/block).
// A fragment (16x32 f16): lane m = lane&15; elem e -> K = hi*16 + e.
// B fragment (32x16): lane n =
// lane&15; elem e -> K = hi*16 + e. C/D: vgpr r, ch = r + 8*hi, col = lane&15.
// ---------------------------------------------------------------------------
template <int K, int O>
__global__ __launch_bounds__(256) void wmma_gemm_bias_stats(
    const _Float16* __restrict__ Wh, const _Float16* __restrict__ X,
    const float* __restrict__ bias, _Float16* __restrict__ Y,
    float* __restrict__ gsum, float* __restrict__ gsq) {
  constexpr int KTS = K >> 5;        // 1 (K=32) or 2 (K=64)
  constexpr int OT  = O >> 4;
  __shared__ float lsum[O], lsq[O];
  const int tid = threadIdx.x;
  if (tid < O) { lsum[tid] = 0.f; lsq[tid] = 0.f; }
  __syncthreads();

  const int lane = tid & 31;
  const int wv   = tid >> 5;
  const int row  = lane & 15;
  const int hi   = lane >> 4;
  const int gw   = blockIdx.x * 8 + wv;
  const int o0   = (gw % OT) << 4;
  const int mg   = gw / OT;          // group of 4 m-tiles (64 columns)

  // A fragments (weights, L2-resident)
  v16h afrag[KTS];
  {
    const _Float16* wrow = Wh + (size_t)(o0 + row) * K;
#pragma unroll
    for (int kt = 0; kt < KTS; ++kt) {
      const _Float16* wp = wrow + (kt << 5);
      afrag[kt] = *(const v16h*)(wp + hi * 16);
    }
  }

  // Bias per lane (8 channels), hoisted
  float bv[8];
#pragma unroll
  for (int r = 0; r < 8; ++r) bv[r] = bias[o0 + (hi << 3) + r];

  // Load all B fragments up front (issues as a vmem clause)
  v16h bfr[4][KTS];
#pragma unroll
  for (int mt = 0; mt < 4; ++mt) {
    const int m0 = (mg * 4 + mt) << 4;
    const _Float16* xp = X + (size_t)(m0 + row) * K + hi * 16;
#pragma unroll
    for (int kt = 0; kt < KTS; ++kt)
      bfr[mt][kt] = *(const v16h*)(xp + (kt << 5));
  }

  // WMMA chain
  v8f acc[4];
#pragma unroll
  for (int mt = 0; mt < 4; ++mt) {
    v8f a = {};
#pragma unroll
    for (int kt = 0; kt < KTS; ++kt)
      a = __builtin_amdgcn_wmma_f32_16x16x32_f16(
          false, afrag[kt], false, bfr[mt][kt], (short)0, a, false, false);
    acc[mt] = a;
  }

  // Epilogue: bias + store + register-accumulated stats
  float ys[8], sq[8];
#pragma unroll
  for (int r = 0; r < 8; ++r) { ys[r] = 0.f; sq[r] = 0.f; }
#pragma unroll
  for (int mt = 0; mt < 4; ++mt) {
    const int m0 = (mg * 4 + mt) << 4;
#pragma unroll
    for (int r = 0; r < 8; ++r) {
      const int ch = o0 + (hi << 3) + r;
      const float y = acc[mt][r] + bv[r];
      Y[(size_t)ch * MTOT + (m0 + row)] = (_Float16)y;
      ys[r] += y;
      sq[r] += y * y;
    }
  }
  // One reduction per wave across the 16 lanes of each half
#pragma unroll
  for (int r = 0; r < 8; ++r) {
#pragma unroll
    for (int off = 1; off < 16; off <<= 1) {   // stays within 16-lane half
      ys[r] += __shfl_xor(ys[r], off);
      sq[r] += __shfl_xor(sq[r], off);
    }
  }
  if (row == 0) {
#pragma unroll
    for (int r = 0; r < 8; ++r) {
      const int ch = o0 + (hi << 3) + r;
      atomicAdd(&lsum[ch], ys[r]);
      atomicAdd(&lsq[ch],  sq[r]);
    }
  }
  __syncthreads();
  if (tid < O) {
    atomicAdd(&gsum[tid], lsum[tid]);
    atomicAdd(&gsq[tid],  lsq[tid]);
  }
}

// ---------------------------------------------------------------------------
// BN finalize: scale = g*rsqrt(var+eps), shift = beta - scale*mean
// ---------------------------------------------------------------------------
__global__ void bn_finalize(const float* __restrict__ gsum,
                            const float* __restrict__ gsq,
                            const float* __restrict__ g,
                            const float* __restrict__ beta,
                            float* __restrict__ scale,
                            float* __restrict__ shift, int O) {
  const int ch = threadIdx.x;
  if (ch < O) {
    const float inv = 1.0f / (float)MTOT;
    const float mean = gsum[ch] * inv;
    const float var  = gsq[ch] * inv - mean * mean;
    const float s    = g[ch] * rsqrtf(var + EPSBN);
    scale[ch] = s;
    shift[ch] = beta[ch] - s * mean;
  }
}

// ---------------------------------------------------------------------------
// Normalize + ReLU + repack to [M][64] f16 (K-contiguous for next GEMM).
// ---------------------------------------------------------------------------
__global__ __launch_bounds__(256) void norm_relu_pack64(
    const _Float16* __restrict__ Y, const float* __restrict__ scale,
    const float* __restrict__ shift, _Float16* __restrict__ Xn) {
  const int m = blockIdx.x * 256 + threadIdx.x;
  v16h* dst = (v16h*)(Xn + (size_t)m * 64);
#pragma unroll
  for (int i = 0; i < 4; ++i) {
    v16h t = {};
#pragma unroll
    for (int e = 0; e < 16; ++e) {
      const int ch = i * 16 + e;
      const float v = (float)Y[(size_t)ch * MTOT + m] * scale[ch] + shift[ch];
      t[e] = (_Float16)fmaxf(v, 0.f);
    }
    dst[i] = t;
  }
}

// ---------------------------------------------------------------------------
// Last layer: normalize + ReLU + max over the 32 samples, write (B,N,128).
// Reads are coalesced (consecutive bn per lane, 64B contiguous each).
// ---------------------------------------------------------------------------
__global__ __launch_bounds__(256) void bn_relu_max_out(
    const _Float16* __restrict__ Y, const float* __restrict__ scale,
    const float* __restrict__ shift, float* __restrict__ out_np) {
  const int o  = blockIdx.y;
  const int bn = blockIdx.x * 256 + threadIdx.x;
  const float s = scale[o], sh = shift[o];
  const v16h* yp = (const v16h*)(Y + (size_t)o * MTOT + (size_t)bn * NS);
  float mx = 0.f;  // max(0, max(v)) == max over s of relu(v)
#pragma unroll
  for (int i = 0; i < 2; ++i) {
    v16h a = yp[i];
#pragma unroll
    for (int e = 0; e < 16; ++e) mx = fmaxf(mx, (float)a[e] * s + sh);
  }
  out_np[(size_t)bn * 128 + o] = mx;
}

// ---------------------------------------------------------------------------
// Small helpers: zero stats, convert/pad weights to f16.
// ---------------------------------------------------------------------------
__global__ void zero_stats(float* __restrict__ p, int n) {
  const int i = blockIdx.x * 256 + threadIdx.x;
  if (i < n) p[i] = 0.f;
}

__global__ void pack_weights(const float* __restrict__ W1,
                             const float* __restrict__ W2,
                             const float* __restrict__ W3,
                             _Float16* __restrict__ Wh1,
                             _Float16* __restrict__ Wh2,
                             _Float16* __restrict__ Wh3) {
  const int t = blockIdx.x * 256 + threadIdx.x;
  if (t < 64 * 32) {                       // W1: 64x9 -> 64x32 zero padded
    const int o = t >> 5, k = t & 31;
    Wh1[t] = (_Float16)((k < 9) ? W1[o * 9 + k] : 0.f);
  }
  if (t < 64 * 64)  Wh2[t] = (_Float16)W2[t];
  if (t < 128 * 64) Wh3[t] = (_Float16)W3[t];
}

// ---------------------------------------------------------------------------
extern "C" void kernel_launch(void* const* d_in, const int* in_sizes, int n_in,
                              void* d_out, int out_size, void* d_ws, size_t ws_size,
                              hipStream_t stream) {
  (void)in_sizes; (void)n_in; (void)out_size; (void)ws_size;
  const float* xyz = (const float*)d_in[0];
  const float* W1  = (const float*)d_in[1];
  const float* b1  = (const float*)d_in[2];
  const float* g1  = (const float*)d_in[3];
  const float* be1 = (const float*)d_in[4];
  const float* W2  = (const float*)d_in[5];
  const float* b2  = (const float*)d_in[6];
  const float* g2  = (const float*)d_in[7];
  const float* be2 = (const float*)d_in[8];
  const float* W3  = (const float*)d_in[9];
  const float* b3  = (const float*)d_in[10];
  const float* g3  = (const float*)d_in[11];
  const float* be3 = (const float*)d_in[12];

  float* out      = (float*)d_out;
  float* out_xyz  = out;                       // 4*4096*3
  float* out_np   = out + (size_t)BATCH * NPT * 3;

  char* ws = (char*)d_ws;
  _Float16* Xbuf = (_Float16*)ws;                          // X ping (<= 64 MiB)
  _Float16* Ybuf = (_Float16*)(ws + (64ull << 20));        // Y pong (<= 128 MiB)
  float* stats   = (float*)(ws + (192ull << 20));
  float* gsum1 = stats;        float* gsq1 = stats + 128;
  float* gsum2 = stats + 256;  float* gsq2 = stats + 384;
  float* gsum3 = stats + 512;  float* gsq3 = stats + 640;
  float* sc1 = stats + 768;    float* sh1 = stats + 896;
  float* sc2 = stats + 1024;   float* sh2 = stats + 1152;
  float* sc3 = stats + 1280;   float* sh3 = stats + 1408;
  _Float16* Wh1 = (_Float16*)(stats + 1536);
  _Float16* Wh2 = Wh1 + 64 * 32;
  _Float16* Wh3 = Wh2 + 64 * 64;

  zero_stats<<<3, 256, 0, stream>>>(stats, 768);
  pack_weights<<<32, 256, 0, stream>>>(W1, W2, W3, Wh1, Wh2, Wh3);

  // Ball query + feature build, one wave per query point (16384 waves).
  ball_query_feats<<<dim3(NPT / 8, BATCH), 256, 0, stream>>>(xyz, Xbuf, out_xyz);

  // Layer 1: 9(->32 padded) -> 64. waves = (64/16)*(MTOT/64) = 32768 -> 4096 blocks
  wmma_gemm_bias_stats<32, 64><<<4096, 256, 0, stream>>>(Wh1, Xbuf, b1, Ybuf, gsum1, gsq1);
  bn_finalize<<<1, 128, 0, stream>>>(gsum1, gsq1, g1, be1, sc1, sh1, 64);
  norm_relu_pack64<<<MTOT / 256, 256, 0, stream>>>(Ybuf, sc1, sh1, Xbuf);

  // Layer 2: 64 -> 64.
  wmma_gemm_bias_stats<64, 64><<<4096, 256, 0, stream>>>(Wh2, Xbuf, b2, Ybuf, gsum2, gsq2);
  bn_finalize<<<1, 128, 0, stream>>>(gsum2, gsq2, g2, be2, sc2, sh2, 64);
  norm_relu_pack64<<<MTOT / 256, 256, 0, stream>>>(Ybuf, sc2, sh2, Xbuf);

  // Layer 3: 64 -> 128. waves = (128/16)*(MTOT/64) = 65536 -> 8192 blocks
  wmma_gemm_bias_stats<64, 128><<<8192, 256, 0, stream>>>(Wh3, Xbuf, b3, Ybuf, gsum3, gsq3);
  bn_finalize<<<1, 128, 0, stream>>>(gsum3, gsq3, g3, be3, sc3, sh3, 128);

  // Fused normalize + ReLU + max over samples -> (B,N,128)
  bn_relu_max_out<<<dim3((BATCH * NPT) / 256, 128), 256, 0, stream>>>(Ybuf, sc3, sh3, out_np);
}
